// GPT_4063039062496
// MI455X (gfx1250) — matrix operand order repaired and checked by reading, not compiled, not run on hardware
//
#include <hip/hip_runtime.h>
#include <hip/hip_bf16.h>
#include <math.h>

// GPT-2 forward, MI455X (gfx1250, wave32, WMMA bf16 + TDM tile staging).
#define BQ   8
#define TT   1024
#define CCH  768
#define HH   12
#define DD   64
#define LL   12
#define VV   50304
#define EPSL 1e-5f

typedef __bf16 bf16;
typedef __attribute__((ext_vector_type(16))) __bf16 v16bf;
typedef __attribute__((ext_vector_type(8)))  __bf16 v8bf;
typedef __attribute__((ext_vector_type(8)))  float  v8f;
typedef __attribute__((ext_vector_type(4)))  unsigned int v4u;
typedef __attribute__((ext_vector_type(8)))  int v8i;
typedef __attribute__((ext_vector_type(4)))  int v4i;

union V16u { v16bf v; v8bf h[2]; };

static __device__ inline v16bf make_frag(v8bf lo, v8bf hi) {
    V16u u; u.h[0] = lo; u.h[1] = hi; return u.v;
}

static __device__ inline float wave_sum(float s) {
    #pragma unroll
    for (int m = 16; m >= 1; m >>= 1) s += __shfl_xor(s, m, 32);
    return s;
}

// ---------------- TDM: 2D tile load global -> LDS (D# per CDNA5 ISA ch.8) ----------------
// tile_w elements per row (bf16), tile_h rows, row stride in elements.
// padc_i/padc_a: pad_interval / pad_amount codes (LDS row padding applied by TDM).
static __device__ __attribute__((always_inline)) void tdm_load_2d(
    unsigned lds_off, const bf16* gptr,
    unsigned tile_w, unsigned tile_h, unsigned stride_elems,
    unsigned padc_i, unsigned padc_a)
{
    unsigned long long ga = (unsigned long long)(size_t)gptr;
    v4u g0;
    g0.x = 1u;                                            // count=1, user mode
    g0.y = lds_off;                                       // lds_addr (group-segment offset)
    g0.z = (unsigned)(ga & 0xFFFFFFFFu);                  // global_addr[31:0]
    g0.w = (unsigned)((ga >> 32) & 0x01FFFFFFu) | (2u << 30);  // global_addr[56:32], type=2
    unsigned w0 = (1u << 16);                             // data_size=1 (2 bytes)
    w0 |= (1u << 20) | (padc_i << 22) | (padc_a << 25);   // pad_enable + codes
    v8i g1;
    g1[0] = (int)w0;
    g1[1] = (int)((tile_w & 0xFFFFu) << 16);              // tensor_dim0[15:0] @ bits[63:48]
    g1[2] = (int)(((tile_w >> 16) & 0xFFFFu) | ((tile_h & 0xFFFFu) << 16)); // dim0 hi | dim1 lo
    g1[3] = (int)(((tile_h >> 16) & 0xFFFFu) | ((tile_w & 0xFFFFu) << 16)); // dim1 hi | tile_dim0
    g1[4] = (int)(tile_h & 0xFFFFu);                      // tile_dim1 (tile_dim2 = 0)
    g1[5] = (int)stride_elems;                            // tensor_dim0_stride[31:0]
    g1[6] = 0;                                            // stride hi / dim1_stride lo
    g1[7] = 0;
    v4i z4 = {0, 0, 0, 0};
    v8i z8 = {0, 0, 0, 0, 0, 0, 0, 0};
    // 6-arg toolchain variant (clang-23 / therock headers)
    __builtin_amdgcn_tensor_load_to_lds(g0, g1, z4, z4, z8, 0);
}

// ---------------- fp32 -> bf16 (plain) ----------------
__global__ void k_f32_to_bf16(const float* __restrict__ src, bf16* __restrict__ dst, size_t n) {
    size_t i = (size_t)blockIdx.x * blockDim.x + threadIdx.x;
    size_t stride = (size_t)gridDim.x * blockDim.x;
    for (; i < n; i += stride) dst[i] = (bf16)src[i];
}

// fp32 [L][K][N] -> bf16 [L][N][K] (N-major weights so B-tiles are row-major block copies)
__global__ void k_convT(const float* __restrict__ src, bf16* __restrict__ dst,
                        int K, int N, size_t total) {
    size_t i = (size_t)blockIdx.x * blockDim.x + threadIdx.x;
    size_t stride = (size_t)gridDim.x * blockDim.x;
    size_t KN = (size_t)K * N;
    for (; i < total; i += stride) {
        size_t layer = i / KN;
        size_t r = i - layer * KN;
        int n = (int)(r / K), k = (int)(r % K);
        dst[i] = (bf16)src[layer * KN + (size_t)k * N + n];
    }
}

// ---------------- embedding: x = lm_w.T[idx] + wpe ----------------
__global__ void k_embed(const int* __restrict__ idx, const float* __restrict__ lm_w,
                        const float* __restrict__ wpe, float* __restrict__ x) {
    int i = blockIdx.x * blockDim.x + threadIdx.x;   // over B*T*C
    int c = i % CCH;
    int bt = i / CCH;
    int t = bt % TT;
    x[i] = lm_w[(size_t)c * VV + idx[bt]] + wpe[t * CCH + c];
}

// ---------------- LayerNorm (center, scale by gamma, no beta), one wave per row ----------------
static __device__ inline void ln_row(const float* __restrict__ in, const float* __restrict__ g,
                                     bf16* __restrict__ o, int lane) {
    float s = 0.f;
    for (int c = lane; c < CCH; c += 32) s += in[c];
    float mean = wave_sum(s) * (1.f / CCH);
    float v = 0.f;
    for (int c = lane; c < CCH; c += 32) { float d = in[c] - mean; v += d * d; }
    float inv = rsqrtf(wave_sum(v) * (1.f / CCH) + EPSL);
    for (int c = lane; c < CCH; c += 32) o[c] = (bf16)((in[c] - mean) * inv * g[c]);
}

__global__ void k_layernorm(const float* __restrict__ x, const float* __restrict__ g,
                            bf16* __restrict__ out, int rows) {
    int wave = (blockIdx.x * blockDim.x + threadIdx.x) >> 5;
    int lane = threadIdx.x & 31;
    if (wave >= rows) return;
    ln_row(x + (size_t)wave * CCH, g, out + (size_t)wave * CCH, lane);
}

// final LN of each batch's last token into padded 128x768 bf16 tile (rows 8..127 zeroed)
__global__ void k_lnf(const float* __restrict__ x, const float* __restrict__ g,
                      bf16* __restrict__ xf) {
    int wave = (blockIdx.x * blockDim.x + threadIdx.x) >> 5;
    int lane = threadIdx.x & 31;
    if (wave >= 128) return;
    bf16* o = xf + (size_t)wave * CCH;
    if (wave >= BQ) {
        for (int c = lane; c < CCH; c += 32) o[c] = (bf16)0.0f;
        return;
    }
    ln_row(x + ((size_t)wave * TT + (TT - 1)) * CCH, g, o, lane);
}

// ---------------- WMMA bf16 GEMM, TDM-staged tiles, fused epilogue ----------------
// A: [M][K] bf16 row-major.  W: [N][K] bf16 (N-major).  out = A @ W^T(N-major) = A @ W_orig.
#define GF_GELU 1
#define GF_RES  2
#define GF_WB   4
#define GF_WF   8
#define GF_BIAS 16

__global__ __launch_bounds__(256) void k_gemm(
    const bf16* __restrict__ A, const bf16* __restrict__ W,
    const float* __restrict__ bias, float* __restrict__ outF,
    bf16* __restrict__ outB, const float* __restrict__ resid,
    int M, int N, int K, int Mvalid, int flags)
{
    __shared__ bf16 As[128 * 40];   // 128x32 tile, TDM-padded rows: 64B data + 16B pad
    __shared__ bf16 Bs[64 * 40];    // 64x32 tile (n-major), same padding

    int t = threadIdx.x;
    int lane = t & 31, wave = t >> 5;
    int wm = (wave >> 1) * 32, wn = (wave & 1) * 32;
    int bn = blockIdx.x * 64, bm = blockIdx.y * 128;
    int r = lane & 15, kh = lane >> 4;

    unsigned asOff = (unsigned)(size_t)As;   // LDS group-segment offset (addr[31:0])
    unsigned bsOff = (unsigned)(size_t)Bs;

    v8f acc[2][2] = {};

    for (int kt = 0; kt < K; kt += 32) {
        if (wave == 0) {
            // rows 64B -> pad_interval code 3 (16 DW), pad 16B -> pad_amount code 3 (4 DW)
            tdm_load_2d(asOff, A + (size_t)bm * K + kt, 32, 128, (unsigned)K, 3, 3);
            tdm_load_2d(bsOff, W + (size_t)bn * K + kt, 32, 64,  (unsigned)K, 3, 3);
        }
        __builtin_amdgcn_s_wait_tensorcnt(0);
        __syncthreads();

        v16bf afr[2], bfr[2];
        #pragma unroll
        for (int i = 0; i < 2; ++i) {
            const bf16* pa = As + (wm + i * 16 + r) * 40;
            afr[i] = make_frag(*(const v8bf*)(pa + kh * 8), *(const v8bf*)(pa + 16 + kh * 8));
        }
        #pragma unroll
        for (int j = 0; j < 2; ++j) {
            const bf16* pb = Bs + (wn + j * 16 + r) * 40 + kh * 8;
            bfr[j] = make_frag(*(const v8bf*)pb, *(const v8bf*)(pb + 16));
        }
        #pragma unroll
        for (int i = 0; i < 2; ++i)
            #pragma unroll
            for (int j = 0; j < 2; ++j)
                acc[i][j] = __builtin_amdgcn_wmma_f32_16x16x32_bf16(
                    false, afr[i], false, bfr[j], (short)0, acc[i][j], false, false);
        __syncthreads();
    }

    // epilogue: bias / GELU(erf) / residual / store f32 and/or bf16
    #pragma unroll
    for (int i = 0; i < 2; ++i) {
        int rowb = bm + wm + i * 16 + kh * 8;
        #pragma unroll
        for (int j = 0; j < 2; ++j) {
            int col = bn + wn + j * 16 + r;
            float bv = (flags & GF_BIAS) ? bias[col] : 0.f;
            #pragma unroll
            for (int rr = 0; rr < 8; ++rr) {
                int row = rowb + rr;
                float v = acc[i][j][rr] + bv;
                if (flags & GF_GELU) v = 0.5f * v * (1.f + erff(v * 0.70710678118654752f));
                if (row < Mvalid) {
                    size_t off = (size_t)row * N + col;
                    if (flags & GF_RES) v += resid[off];
                    if (flags & GF_WF) outF[off] = v;
                    if (flags & GF_WB) outB[off] = (bf16)v;
                }
            }
        }
    }
}

// ---------------- causal flash attention (bf16 WMMA, online softmax, TDM K-tiles) ----------------
__global__ __launch_bounds__(128) void k_attn(const bf16* __restrict__ qkv,
                                              bf16* __restrict__ attout)
{
    __shared__ bf16 Kt[32][72];       // [key][d] TDM-padded: 128B data + 16B pad
    __shared__ bf16 Vt[64][40];       // [d][key] transposed (manual staging)
    __shared__ bf16 Pb[4][16][40];    // per-wave probability tile 16x32

    int t = threadIdx.x, lane = t & 31, wave = t >> 5;
    int blk = blockIdx.x;             // ((b*H + h)*(T/64) + qt)
    int qt = blk % (TT / 64); blk /= (TT / 64);
    int h = blk % HH; int b = blk / HH;
    int qbase = qt * 64 + wave * 16;

    const int QS = 3 * CCH;
    size_t baseRow = (size_t)b * TT;
    int r = lane & 15, kh = lane >> 4;
    unsigned ktOff = (unsigned)(size_t)&Kt[0][0];

    // preload this wave's 16x64 Q block as two A-fragments (d 0..31, 32..63)
    const bf16* qrow = qkv + (baseRow + qbase + r) * QS + h * DD;
    v16bf aq[2];
    #pragma unroll
    for (int d = 0; d < 2; ++d)
        aq[d] = make_frag(*(const v8bf*)(qrow + d * 32 + kh * 8),
                          *(const v8bf*)(qrow + d * 32 + 16 + kh * 8));

    v8f o[4] = {};
    float mI[8], lI[8];
    #pragma unroll
    for (int i = 0; i < 8; ++i) { mI[i] = -__builtin_inff(); lI[i] = 0.f; }

    int nk = qt * 64 + 64;            // keys needed by this block (causal)
    for (int kb = 0; kb < nk; kb += 32) {
        // K tile via TDM: 32 keys x 64 d, stride 3C; 128B rows -> codes (4, 3)
        if (wave == 0)
            tdm_load_2d(ktOff, qkv + (baseRow + kb) * QS + CCH + h * DD,
                        64, 32, (unsigned)QS, 4, 3);
        // V tile staged transposed by all 128 threads
        {
            int key = t >> 2, d0 = (t & 3) * 16;
            const bf16* vr = qkv + (baseRow + kb + key) * QS + 2 * CCH + h * DD + d0;
            v8bf v0 = *(const v8bf*)vr, v1 = *(const v8bf*)(vr + 8);
            #pragma unroll
            for (int z = 0; z < 8; ++z) {
                Vt[d0 + z][key]     = v0[z];
                Vt[d0 + 8 + z][key] = v1[z];
            }
            if (kb + 32 < nk) __builtin_prefetch(vr + 32 * QS, 0, 3);
        }
        __builtin_amdgcn_s_wait_tensorcnt(0);
        __syncthreads();

        // S = Q @ K^T for 16 q-rows x 32 keys (two 16-key column tiles)
        v8f s0 = {}, s1 = {};
        #pragma unroll
        for (int d = 0; d < 2; ++d) {
            const bf16* p0 = &Kt[r][d * 32 + kh * 8];
            const bf16* p1 = &Kt[16 + r][d * 32 + kh * 8];
            v16bf bk0 = make_frag(*(const v8bf*)p0, *(const v8bf*)(p0 + 16));
            v16bf bk1 = make_frag(*(const v8bf*)p1, *(const v8bf*)(p1 + 16));
            s0 = __builtin_amdgcn_wmma_f32_16x16x32_bf16(false, aq[d], false, bk0, (short)0, s0, false, false);
            s1 = __builtin_amdgcn_wmma_f32_16x16x32_bf16(false, aq[d], false, bk1, (short)0, s1, false, false);
        }

        // causal mask, scale, online softmax (row reductions across 16-lane halves)
        #pragma unroll
        for (int rr = 0; rr < 8; ++rr) {
            int q  = qbase + rr + kh * 8;
            int c0 = kb + r, c1 = kb + 16 + r;
            float x0 = (c0 <= q) ? s0[rr] * 0.125f : -__builtin_inff();
            float x1 = (c1 <= q) ? s1[rr] * 0.125f : -__builtin_inff();
            float mx = fmaxf(x0, x1);
            #pragma unroll
            for (int m = 8; m >= 1; m >>= 1) mx = fmaxf(mx, __shfl_xor(mx, m, 32));
            float mnew  = fmaxf(mI[rr], mx);
            float scale = expf(mI[rr] - mnew);
            float p0 = expf(x0 - mnew);
            float p1 = expf(x1 - mnew);
            float ps = p0 + p1;
            #pragma unroll
            for (int m = 8; m >= 1; m >>= 1) ps += __shfl_xor(ps, m, 32);
            lI[rr] = lI[rr] * scale + ps;
            mI[rr] = mnew;
            #pragma unroll
            for (int j = 0; j < 4; ++j) o[j][rr] *= scale;
            int ql = rr + kh * 8;
            Pb[wave][ql][r]      = (bf16)p0;
            Pb[wave][ql][16 + r] = (bf16)p1;
        }

        // O += P @ V  (P via per-wave LDS round-trip into A-fragment layout)
        {
            const bf16* pp = &Pb[wave][r][0];
            v16bf pf = make_frag(*(const v8bf*)(pp + kh * 8), *(const v8bf*)(pp + 16 + kh * 8));
            #pragma unroll
            for (int j = 0; j < 4; ++j) {
                const bf16* pv = &Vt[j * 16 + r][kh * 8];
                v16bf vf = make_frag(*(const v8bf*)pv, *(const v8bf*)(pv + 16));
                o[j] = __builtin_amdgcn_wmma_f32_16x16x32_bf16(false, pf, false, vf, (short)0, o[j], false, false);
            }
        }
        __syncthreads();
    }

    // normalize and write bf16 head output
    #pragma unroll
    for (int j = 0; j < 4; ++j)
        #pragma unroll
        for (int rr = 0; rr < 8; ++rr) {
            int q = qbase + rr + kh * 8;
            int col = h * DD + j * 16 + r;
            attout[(baseRow + q) * (size_t)CCH + col] = (bf16)(o[j][rr] / lI[rr]);
        }
}

// ---------------- host-side orchestration ----------------
extern "C" void kernel_launch(void* const* d_in, const int* in_sizes, int n_in,
                              void* d_out, int out_size, void* d_ws, size_t ws_size,
                              hipStream_t stream) {
    const int*   idx     = (const int*)d_in[0];
    const float* lm_w    = (const float*)d_in[1];
    const float* wpe     = (const float*)d_in[2];
    const float* attn_w  = (const float*)d_in[3];
    const float* attn_b  = (const float*)d_in[4];
    const float* proj_w  = (const float*)d_in[5];
    const float* proj_b  = (const float*)d_in[6];
    const float* ln1_g   = (const float*)d_in[7];
    const float* ln2_g   = (const float*)d_in[8];
    const float* fc_w    = (const float*)d_in[9];
    const float* fc_b    = (const float*)d_in[10];
    const float* mproj_w = (const float*)d_in[11];
    const float* mproj_b = (const float*)d_in[12];
    const float* lnf_g   = (const float*)d_in[13];

    const size_t M = (size_t)BQ * TT;   // 8192 tokens
    char* ws = (char*)d_ws;
    size_t off = 0;
    auto alloc = [&](size_t bytes) -> void* {
        void* p = ws + off;
        off += (bytes + 255) & ~(size_t)255;
        return p;
    };

    float* x      = (float*)alloc(M * CCH * sizeof(float));
    bf16*  xn     = (bf16*) alloc(M * CCH * sizeof(bf16));
    bf16*  qkvb   = (bf16*) alloc(M * 3 * CCH * sizeof(bf16));
    bf16*  atto   = (bf16*) alloc(M * CCH * sizeof(bf16));
    bf16*  hbuf   = (bf16*) alloc(M * 4 * CCH * sizeof(bf16));
    bf16*  xf     = (bf16*) alloc((size_t)128 * CCH * sizeof(bf16));
    // all weights stored N-major ([N][K]) for TDM-friendly row-major B tiles
    bf16*  wbA    = (bf16*) alloc((size_t)LL * CCH * 3 * CCH * sizeof(bf16));
    bf16*  wbP    = (bf16*) alloc((size_t)LL * CCH * CCH * sizeof(bf16));
    bf16*  wbF    = (bf16*) alloc((size_t)LL * CCH * 4 * CCH * sizeof(bf16));
    bf16*  wbM    = (bf16*) alloc((size_t)LL * 4 * CCH * CCH * sizeof(bf16));
    bf16*  wbLM   = (bf16*) alloc((size_t)CCH * VV * sizeof(bf16));

    auto convT = [&](const float* s, bf16* d, int K, int N, size_t total) {
        k_convT<<<4096, 256, 0, stream>>>(s, d, K, N, total);
    };
    convT(attn_w,  wbA,  CCH,     3 * CCH, (size_t)LL * CCH * 3 * CCH);
    convT(proj_w,  wbP,  CCH,     CCH,     (size_t)LL * CCH * CCH);
    convT(fc_w,    wbF,  CCH,     4 * CCH, (size_t)LL * CCH * 4 * CCH);
    convT(mproj_w, wbM,  4 * CCH, CCH,     (size_t)LL * 4 * CCH * CCH);
    convT(lm_w,    wbLM, CCH,     VV,      (size_t)CCH * VV);

    k_embed<<<(int)(M * CCH / 256), 256, 0, stream>>>(idx, lm_w, wpe, x);

    for (int l = 0; l < LL; ++l) {
        // attention half
        k_layernorm<<<(int)(M / 8), 256, 0, stream>>>(x, ln1_g + (size_t)l * CCH, xn, (int)M);
        k_gemm<<<dim3(3 * CCH / 64, (unsigned)(M / 128)), 256, 0, stream>>>(
            xn, wbA + (size_t)l * CCH * 3 * CCH, attn_b + (size_t)l * 3 * CCH,
            nullptr, qkvb, nullptr, (int)M, 3 * CCH, CCH, (int)M, GF_WB | GF_BIAS);
        k_attn<<<BQ * HH * (TT / 64), 128, 0, stream>>>(qkvb, atto);
        k_gemm<<<dim3(CCH / 64, (unsigned)(M / 128)), 256, 0, stream>>>(
            atto, wbP + (size_t)l * CCH * CCH, proj_b + (size_t)l * CCH,
            x, nullptr, x, (int)M, CCH, CCH, (int)M, GF_WF | GF_RES | GF_BIAS);
        // MLP half
        k_layernorm<<<(int)(M / 8), 256, 0, stream>>>(x, ln2_g + (size_t)l * CCH, xn, (int)M);
        k_gemm<<<dim3(4 * CCH / 64, (unsigned)(M / 128)), 256, 0, stream>>>(
            xn, wbF + (size_t)l * CCH * 4 * CCH, fc_b + (size_t)l * 4 * CCH,
            nullptr, hbuf, nullptr, (int)M, 4 * CCH, CCH, (int)M, GF_WB | GF_BIAS | GF_GELU);
        k_gemm<<<dim3(CCH / 64, (unsigned)(M / 128)), 256, 0, stream>>>(
            hbuf, wbM + (size_t)l * 4 * CCH * CCH, mproj_b + (size_t)l * CCH,
            x, nullptr, x, (int)M, CCH, 4 * CCH, (int)M, GF_WF | GF_RES | GF_BIAS);
    }

    // final LN of last tokens + logits GEMM (M padded to 128, only 8 rows stored)
    k_lnf<<<16, 256, 0, stream>>>(x, lnf_g, xf);
    k_gemm<<<dim3(VV / 64, 1), 256, 0, stream>>>(
        xf, wbLM, nullptr, (float*)d_out, nullptr, nullptr,
        128, VV, CCH, BQ, GF_WF);
}
